// EmbracementLayer_51642686768041
// MI455X (gfx1250) — matrix operand order repaired and checked by reading, not compiled, not run on hardware
//
#include <hip/hip_runtime.h>

// Problem sizes (fixed by the reference)
#define Bsz 16
#define Ssz 512
#define Esz 1024
#define Hsz 16
#define Dsz 64
#define Tsz (Bsz * Ssz) // 8192 token rows

typedef __attribute__((ext_vector_type(16))) __bf16 v16bf;
typedef __attribute__((ext_vector_type(8)))  float  v8f;

// dword-granular fragment builder: each 32-bit slot holds bf16 pair
// (element 2v in [15:0], 2v+1 in [31:16]) matching ISA 7.12.2 layouts.
union FragBF { v16bf v; unsigned int u[8]; };

static __device__ __forceinline__ unsigned short f32_to_bf16_bits(float f) {
  unsigned int u = __float_as_uint(f);
  unsigned int lsb = (u >> 16) & 1u;
  u += 0x7fffu + lsb; // round-to-nearest-even
  return (unsigned short)(u >> 16);
}

// ---- WMMA fragment loaders (contiguous-k order k = 16*hf + s, wave32) -------------------
// A (16x32 bf16, MxK), row-major A[m][k]:
//   lane = hf*16 + m ; VGPR v holds k-pair hf*16 + 2v
static __device__ __forceinline__ v16bf load_a_frag(const unsigned short* tile,
                                                    int stride, int lane) {
  const int hf = lane >> 4, m = lane & 15;
  const unsigned short* rp = tile + m * stride;
  FragBF f;
#pragma unroll
  for (int v = 0; v < 8; ++v) {
    const int k2 = hf * 16 + 2 * v;
    f.u[v] = *(const unsigned int*)(rp + k2); // 4B-aligned pair
  }
  return f.v;
}

// B (32x16 bf16, KxN) with memory holding Bt[n][k] (n-major rows, contiguous
// along k — i.e. the natural layout of W[n][k] and of key rows K[s][d]):
//   lane = hf*16 + n ; VGPR v holds k-pair hf*16 + 2v
static __device__ __forceinline__ v16bf load_b_frag_nmajor(const unsigned short* tile,
                                                           int stride, int lane) {
  const int hf = lane >> 4, n = lane & 15;
  const unsigned short* rp = tile + n * stride + hf * 16;
  FragBF f;
#pragma unroll
  for (int v = 0; v < 8; ++v) f.u[v] = *(const unsigned int*)(rp + 2 * v);
  return f.v;
}

// ---- fp32 -> bf16 bulk convert (tokens and weights; no transpose needed) --
__global__ void k_f32_to_bf16(const float* __restrict__ src,
                              unsigned short* __restrict__ dst, int n4) {
  const int i = blockIdx.x * blockDim.x + threadIdx.x;
  if (i < n4) {
    const float4 v = ((const float4*)src)[i];
    ushort4 o;
    o.x = f32_to_bf16_bits(v.x); o.y = f32_to_bf16_bits(v.y);
    o.z = f32_to_bf16_bits(v.z); o.w = f32_to_bf16_bits(v.w);
    ((ushort4*)dst)[i] = o;
  }
}

// ---- Projection GEMM: Y[t][n] = sum_k X[t][k]*W[n][k] + bias[n] -----------
// 64x64 block tile, K-step 32. Both LDS tiles are staged row-major with
// contiguous K (fully coalesced b128 both directions); 8 waves each own a
// 16x32 output slab (1 shared A frag, 2 WMMAs). Output goes straight to
// [B,H,S,D] bf16.
__global__ __launch_bounds__(256) void k_proj_gemm(
    const unsigned short* __restrict__ Xbf,  // [T][E] bf16
    const unsigned short* __restrict__ Wbf,  // [E][E] bf16, W[n][k]
    const float* __restrict__ bias,          // [E] fp32
    unsigned short* __restrict__ Ybf)        // [B,H,S,D] bf16
{
  __shared__ __align__(16) unsigned short As[64 * 32];
  __shared__ __align__(16) unsigned short Bs[64 * 32];
  const int tid = threadIdx.x;
  const int lane = tid & 31, w = tid >> 5;
  const int wm = w & 3, wn = w >> 2;
  const int t0 = blockIdx.x * 64;
  const int n0 = blockIdx.y * 64;

  v8f acc0 = {}; v8f acc1 = {};

  const int r8 = tid >> 2, seg = tid & 3; // 64 rows x 32 cols, 16B per thread

  for (int k0 = 0; k0 < Esz; k0 += 32) {
    *((uint4*)&As[r8 * 32 + seg * 8]) =
        *((const uint4*)&Xbf[(size_t)(t0 + r8) * Esz + k0 + seg * 8]);
    *((uint4*)&Bs[r8 * 32 + seg * 8]) =
        *((const uint4*)&Wbf[(size_t)(n0 + r8) * Esz + k0 + seg * 8]);
    if (k0 + 32 < Esz) { // hide next slice's HBM latency behind the WMMAs
      __builtin_prefetch(&Xbf[(size_t)(t0 + r8) * Esz + k0 + 32 + seg * 8], 0, 0);
      __builtin_prefetch(&Wbf[(size_t)(n0 + r8) * Esz + k0 + 32 + seg * 8], 0, 0);
    }
    __syncthreads();
    const v16bf a  = load_a_frag(&As[wm * 16 * 32], 32, lane);
    const v16bf b0 = load_b_frag_nmajor(&Bs[(wn * 2 + 0) * 16 * 32], 32, lane);
    const v16bf b1 = load_b_frag_nmajor(&Bs[(wn * 2 + 1) * 16 * 32], 32, lane);
    acc0 = __builtin_amdgcn_wmma_f32_16x16x32_bf16(false, a, false, b0,
                                                   (short)0, acc0, false, false);
    acc1 = __builtin_amdgcn_wmma_f32_16x16x32_bf16(false, a, false, b1,
                                                   (short)0, acc1, false, false);
    __syncthreads();
  }

  const int hf = lane >> 4, nl = lane & 15;
#pragma unroll
  for (int sub = 0; sub < 2; ++sub) {
    const int e_out = n0 + (wn * 2 + sub) * 16 + nl;
    const float bv = bias[e_out];
    const int h = e_out >> 6, d = e_out & 63;
    const v8f acc = sub ? acc1 : acc0;
#pragma unroll
    for (int r = 0; r < 8; ++r) {
      const int m = r + 8 * hf;                 // C/D row layout
      const int t = t0 + wm * 16 + m;
      const int b = t >> 9, s = t & 511;
      Ybf[(((size_t)(b * Hsz + h) * Ssz + s) * Dsz) + d] =
          f32_to_bf16_bits(acc[r] + bv);
    }
  }
}

// ---- Attention: per (b,h,16-query strip) compute 16x512 scores via WMMA,
// softmax rows in LDS, atomically accumulate column sums into col[B,S]. -----
__global__ __launch_bounds__(256) void k_attn(
    const unsigned short* __restrict__ Qbf,  // [B*H][S][D] bf16
    const unsigned short* __restrict__ Kbf,  // [B*H][S][D] bf16
    float* __restrict__ col)                 // [B][S] fp32 (pre-zeroed)
{
  __shared__ float sc[16 * 512];   // 32 KB score strip
  __shared__ float red[16][17];
  __shared__ float rmax[16];
  __shared__ float rinv[16];

  const int tid = threadIdx.x;
  const int lane = tid & 31, w = tid >> 5;
  const int qt = blockIdx.x;                 // 0..31
  const int h = blockIdx.y, b = blockIdx.z;
  const int bh = b * Hsz + h;
  const int q0 = qt * 16;

  const unsigned short* Qbase = Qbf + (size_t)bh * Ssz * Dsz + (size_t)q0 * Dsz;
  const unsigned short* Kbase = Kbf + (size_t)bh * Ssz * Dsz;

  // Q fragment reused across all 4 key tiles this wave owns (D=64 -> 2 K-steps)
  const v16bf a0 = load_a_frag(Qbase + 0,  Dsz, lane);
  const v16bf a1 = load_a_frag(Qbase + 32, Dsz, lane);

  for (int kt = w * 4; kt < w * 4 + 4; ++kt) {
    const unsigned short* Kt = Kbase + (size_t)(kt * 16) * Dsz;
    const v16bf b0 = load_b_frag_nmajor(Kt + 0,  Dsz, lane);
    const v16bf b1 = load_b_frag_nmajor(Kt + 32, Dsz, lane);
    v8f acc = {};
    acc = __builtin_amdgcn_wmma_f32_16x16x32_bf16(false, a0, false, b0,
                                                  (short)0, acc, false, false);
    acc = __builtin_amdgcn_wmma_f32_16x16x32_bf16(false, a1, false, b1,
                                                  (short)0, acc, false, false);
    const int hf = lane >> 4, nl = lane & 15;
    const int c = kt * 16 + nl;
#pragma unroll
    for (int r = 0; r < 8; ++r)
      sc[(r + 8 * hf) * 512 + c] = acc[r] * 0.125f + 1.0f; // /sqrt(64) + mask
  }
  __syncthreads();

  // Row softmax: 16 threads per row, 32 cols each.
  const int row = tid >> 4, part = tid & 15;
  float lmax = -3.4e38f;
  for (int c = part * 32; c < part * 32 + 32; ++c)
    lmax = fmaxf(lmax, sc[row * 512 + c]);
  red[row][part] = lmax;
  __syncthreads();
  if (part == 0) {
    float m = red[row][0];
    for (int i = 1; i < 16; ++i) m = fmaxf(m, red[row][i]);
    rmax[row] = m;
  }
  __syncthreads();
  const float mrow = rmax[row];
  float lsum = 0.f;
  for (int c = part * 32; c < part * 32 + 32; ++c) {
    const float e = __expf(sc[row * 512 + c] - mrow);
    sc[row * 512 + c] = e;
    lsum += e;
  }
  red[row][part] = lsum;
  __syncthreads();
  if (part == 0) {
    float s = 0.f;
    for (int i = 0; i < 16; ++i) s += red[row][i];
    rinv[row] = 1.0f / s;
  }
  __syncthreads();

  // Column sums over the 16 query rows of this strip.
  for (int c = tid; c < 512; c += 256) {
    float s = 0.f;
#pragma unroll
    for (int m = 0; m < 16; ++m) s += sc[m * 512 + c] * rinv[m];
    atomicAdd(&col[b * Ssz + c], s);
  }
}

__global__ void k_zero(float* __restrict__ p, int n) {
  const int i = blockIdx.x * blockDim.x + threadIdx.x;
  if (i < n) p[i] = 0.f;
}

// ---- Per-batch categorical sampling (inverse CDF) + gather ----------------
__global__ __launch_bounds__(512) void k_sample(
    const float* __restrict__ tokens,  // [B][S][E] fp32
    const float* __restrict__ col,     // [B][S]
    float* __restrict__ out)           // [B][E]
{
  __shared__ float cdf[512];
  __shared__ float tmp[512];
  const int tid = threadIdx.x;
  const int b = blockIdx.x;

  const float x = col[b * Ssz + tid];
  tmp[tid] = x;
  __syncthreads();
  for (int off = 256; off > 0; off >>= 1) {
    if (tid < off) tmp[tid] = fmaxf(tmp[tid], tmp[tid + off]);
    __syncthreads();
  }
  const float e = __expf(x - tmp[0]);
  cdf[tid] = e;
  __syncthreads();
  // Hillis-Steele inclusive prefix sum over 512 elements.
  for (int off = 1; off < 512; off <<= 1) {
    const float add = (tid >= off) ? cdf[tid - off] : 0.f;
    __syncthreads();
    cdf[tid] += add;
    __syncthreads();
  }
  const float total = cdf[511];

  for (int j = tid; j < Esz; j += 512) {
    unsigned int s = (unsigned int)(b * Esz + j) * 0x9E3779B9u + 42u * 0x85EBCA6Bu;
    s ^= s >> 16; s *= 0x7FEB352Du; s ^= s >> 15; s *= 0x846CA68Bu; s ^= s >> 16;
    const float u = (float)(s >> 8) * (1.0f / 16777216.0f);
    const float target = u * total;
    int lo = 0, hi = 511;
    while (lo < hi) {
      const int mid = (lo + hi) >> 1;
      if (cdf[mid] < target) lo = mid + 1; else hi = mid;
    }
    out[b * Esz + j] = tokens[((size_t)b * Ssz + lo) * Esz + j];
  }
}

extern "C" void kernel_launch(void* const* d_in, const int* in_sizes, int n_in,
                              void* d_out, int out_size, void* d_ws, size_t ws_size,
                              hipStream_t stream) {
  (void)in_sizes; (void)n_in; (void)out_size; (void)ws_size;
  const float* tokens = (const float*)d_in[0];
  const float* Wq = (const float*)d_in[1];
  const float* bq = (const float*)d_in[2];
  const float* Wk = (const float*)d_in[3];
  const float* bk = (const float*)d_in[4];
  // d_in[5]=Wv, d_in[6]=bv: reference computes v then deletes it -> skip.

  char* ws = (char*)d_ws;
  unsigned short* Xbf  = (unsigned short*)(ws);                        // 16 MB
  unsigned short* Wqbf = (unsigned short*)(ws + (size_t)(16u << 20));  //  2 MB
  unsigned short* Wkbf = (unsigned short*)(ws + (size_t)(18u << 20));  //  2 MB
  unsigned short* Qbf  = (unsigned short*)(ws + (size_t)(20u << 20));  // 16 MB
  unsigned short* Kbf  = (unsigned short*)(ws + (size_t)(36u << 20));  // 16 MB
  float* col           = (float*)(ws + (size_t)(52u << 20));           // 32 KB

  // 1. precision demotion for the matrix-core path (weights keep their
  //    natural n-major layout, which is exactly the WMMA B-operand layout)
  k_f32_to_bf16<<<(Tsz * Esz / 4 + 255) / 256, 256, 0, stream>>>(
      tokens, Xbf, Tsz * Esz / 4);
  k_f32_to_bf16<<<(Esz * Esz / 4 + 255) / 256, 256, 0, stream>>>(
      Wq, Wqbf, Esz * Esz / 4);
  k_f32_to_bf16<<<(Esz * Esz / 4 + 255) / 256, 256, 0, stream>>>(
      Wk, Wkbf, Esz * Esz / 4);

  // 2. Q/K projections (bf16 WMMA, fp32 accum, bias fused, [B,H,S,D] output)
  dim3 gg(Tsz / 64, Esz / 64);
  k_proj_gemm<<<gg, 256, 0, stream>>>(Xbf, Wqbf, bq, Qbf);
  k_proj_gemm<<<gg, 256, 0, stream>>>(Xbf, Wkbf, bk, Kbf);

  // 3. attention score strip + softmax + column-sum accumulation
  k_zero<<<(Bsz * Ssz + 255) / 256, 256, 0, stream>>>(col, Bsz * Ssz);
  dim3 ag(Ssz / 16, Hsz, Bsz);
  k_attn<<<ag, 256, 0, stream>>>(Qbf, Kbf, col);

  // 4. categorical sample + gather
  k_sample<<<Bsz, 512, 0, stream>>>(tokens, col, (float*)d_out);
}
